// Encoder_78649441124984
// MI455X (gfx1250) — matrix operand order repaired and checked by reading, not compiled, not run on hardware
//
#include <hip/hip_runtime.h>
#include <hip/hip_bf16.h>
#include <stdint.h>

// ---------------------------------------------------------------------------
// Encoder block on gfx1250: all contractions via v_wmma_f32_16x16x32_f16.
// Shapes: B=32 T=512 D=500 H=10 HD=50, M=B*T=16384, Kpad=512, HDpad=64.
// GEMM wave tile: 32(M) x 64(N)  -> 8 WMMA per 32-k-step, B frags reused x2.
// ---------------------------------------------------------------------------

#define Bn   32
#define Tn   512
#define Dn   500
#define Hn   10
#define HDn  50
#define Mn   (Bn * Tn)        // 16384
#define KP   512              // padded K for all D-contractions
#define HDP  64               // padded head dim
#define BH   (Bn * Hn)        // 320

typedef __attribute__((ext_vector_type(16))) _Float16 v16h;
typedef __attribute__((ext_vector_type(8)))  _Float16 v8h;
typedef __attribute__((ext_vector_type(8)))  float    v8f;

// ---------------- fragment helpers (contiguous k order, shared by A and B) ----------

// A fragment (16x32 f16): lane holds row m = lane%16; halves [0..7] = A[m, base..base+7],
// halves [8..15] = A[m, base+8..base+15], base = k0 + 16*(lane>=16).
static __device__ inline v16h load_a_frag(const _Float16* __restrict__ row,
                                          int k0, int half16) {
    const int base = k0 + half16 * 16;
    v8h lo = *(const v8h*)(row + base);
    v8h hi = *(const v8h*)(row + base + 8);
    v16h a;
    #pragma unroll
    for (int i = 0; i < 8; ++i) { a[i] = lo[i]; a[i + 8] = hi[i]; }
    return a;
}

// B fragment (32x16 f16) from [N,K] row-major weights: lane holds col n = lane%16,
// halves e = W[n, k0 + e + 16*(lane>=16)]  -> one contiguous 32B load.
static __device__ inline v16h load_b_frag(const _Float16* __restrict__ W,
                                          size_t ldb, int n, int k0, int half16) {
    return *(const v16h*)(W + (size_t)n * ldb + k0 + half16 * 16);
}

static __device__ inline v8f wmma_f16(v16h a, v16h b, v8f c) {
    return __builtin_amdgcn_wmma_f32_16x16x32_f16(false, a, false, b,
                                                  (short)0, c, false, false);
}

static __device__ inline float hmax16(float v) {
    #pragma unroll
    for (int m = 1; m < 16; m <<= 1) v = fmaxf(v, __shfl_xor(v, m, 32));
    return v;
}
static __device__ inline float hsum16(float v) {
    #pragma unroll
    for (int m = 1; m < 16; m <<= 1) v += __shfl_xor(v, m, 32);
    return v;
}

// ---------------- elementwise kernels --------------------------------------

__global__ void k_zero_words(uint32_t* __restrict__ p, size_t nwords) {
    size_t i = (size_t)blockIdx.x * blockDim.x + threadIdx.x;
    size_t stride = (size_t)gridDim.x * blockDim.x;
    for (; i < nwords; i += stride) p[i] = 0u;
}

// pack f32 [N,500] row-major -> f16 [Npad,512] (pad zone pre-zeroed)
__global__ void k_pack_w(const float* __restrict__ src, _Float16* __restrict__ dst, int N) {
    int idx = blockIdx.x * blockDim.x + threadIdx.x;
    if (idx >= N * KP) return;
    int n = idx >> 9, k = idx & (KP - 1);
    if (k < Dn) dst[idx] = (_Float16)src[n * Dn + k];
}

// sinusoidal pos-emb (+x): out[b,t,d] f32
__global__ void k_posemb(const int* __restrict__ ori, const float* __restrict__ x,
                         float* __restrict__ out) {
    size_t i = (size_t)blockIdx.x * blockDim.x + threadIdx.x;
    if (i >= (size_t)Mn * Dn) return;
    int d = (int)(i % Dn);
    size_t m = i / Dn;
    int t = (int)(m & (Tn - 1));
    int b = (int)(m >> 9);
    int tok = ori[b * Tn + t];
    int pos = (tok != 0) ? (t + 1) : 0;
    float pe = 0.f;
    if (pos != 0) {
        int j = (d < Dn / 2) ? d : d - Dn / 2;
        float inv = __expf((float)j * (-9.210340371976184f / (float)(Dn / 2 - 1)));
        float ang = (float)pos * inv;
        pe = (d < Dn / 2) ? __sinf(ang) : __cosf(ang);
    }
    out[i] = pe + x[i];
}

// depthwise conv1d k=7 pad=3 + bias; f32 in [B,T,500] -> f16 out [M,512]
__global__ void k_dwconv(const float* __restrict__ X, const float* __restrict__ dw,
                         const float* __restrict__ db, _Float16* __restrict__ Y) {
    size_t i = (size_t)blockIdx.x * blockDim.x + threadIdx.x;
    if (i >= (size_t)Mn * Dn) return;
    int d = (int)(i % Dn);
    size_t m = i / Dn;
    int t = (int)(m & (Tn - 1));
    size_t b = m >> 9;
    float s = db[d];
    #pragma unroll
    for (int k = 0; k < 7; ++k) {
        int tt = t + k - 3;
        if (tt >= 0 && tt < Tn)
            s += X[(b * Tn + tt) * Dn + d] * dw[d * 7 + k];
    }
    Y[m * KP + d] = (_Float16)s;
}

// ---------------- generic WMMA GEMM: out f32 = A*W^T + bias + res ----------
// A: [M,512] f16, W: [Npad,512] f16. Block tile 128x64, wave tile 32x64.
__global__ __launch_bounds__(128)
void k_gemm_f32out(const _Float16* __restrict__ A, const _Float16* __restrict__ W,
                   const float* __restrict__ bias, const float* __restrict__ res,
                   float* __restrict__ out, _Float16* __restrict__ outH, int Nout) {
    const int lane = threadIdx.x & 31, wave = threadIdx.x >> 5;
    const int half16 = lane >> 4, l16 = lane & 15;
    const int m0 = blockIdx.x * 128 + wave * 32;
    const int n0 = blockIdx.y * 64;
    v8f acc[2][4];
    #pragma unroll
    for (int a = 0; a < 2; ++a)
        #pragma unroll
        for (int j = 0; j < 4; ++j)
            #pragma unroll
            for (int r = 0; r < 8; ++r) acc[a][j][r] = 0.f;
    const _Float16* Arow0 = A + (size_t)(m0 + l16) * KP;
    const _Float16* Arow1 = Arow0 + (size_t)16 * KP;
    for (int k0 = 0; k0 < KP; k0 += 32) {
        if (k0 + 64 < KP) {
            __builtin_prefetch((const void*)(Arow0 + k0 + 64), 0, 3);
            __builtin_prefetch((const void*)(Arow1 + k0 + 64), 0, 3);
        }
        v16h af0 = load_a_frag(Arow0, k0, half16);
        v16h af1 = load_a_frag(Arow1, k0, half16);
        #pragma unroll
        for (int j = 0; j < 4; ++j) {
            v16h bf = load_b_frag(W, KP, n0 + j * 16 + l16, k0, half16);
            acc[0][j] = wmma_f16(af0, bf, acc[0][j]);
            acc[1][j] = wmma_f16(af1, bf, acc[1][j]);
        }
    }
    #pragma unroll
    for (int a = 0; a < 2; ++a)
        #pragma unroll
        for (int j = 0; j < 4; ++j) {
            int n = n0 + j * 16 + l16;
            if (n >= Nout) continue;
            #pragma unroll
            for (int r = 0; r < 8; ++r) {
                int m = m0 + a * 16 + r + 8 * half16;
                float v = acc[a][j][r] + bias[n];
                if (res) v += res[(size_t)m * Nout + n];
                out[(size_t)m * Nout + n] = v;
                if (outH) outH[(size_t)m * KP + n] = (_Float16)v;
            }
        }
}

// ---------------- QKV GEMM: scatters into attention-friendly layouts -------
// W = in_w packed [1536,512]; q -> qh [BH,512,64] (scaled), k -> kh, v -> vth [BH,64,512]
__global__ __launch_bounds__(128)
void k_gemm_qkv(const _Float16* __restrict__ A, const _Float16* __restrict__ W,
                const float* __restrict__ bias, _Float16* __restrict__ qh,
                _Float16* __restrict__ kh, _Float16* __restrict__ vth) {
    const int lane = threadIdx.x & 31, wave = threadIdx.x >> 5;
    const int half16 = lane >> 4, l16 = lane & 15;
    const int m0 = blockIdx.x * 128 + wave * 32;
    const int n0 = blockIdx.y * 64;
    v8f acc[2][4];
    #pragma unroll
    for (int a = 0; a < 2; ++a)
        #pragma unroll
        for (int j = 0; j < 4; ++j)
            #pragma unroll
            for (int r = 0; r < 8; ++r) acc[a][j][r] = 0.f;
    const _Float16* Arow0 = A + (size_t)(m0 + l16) * KP;
    const _Float16* Arow1 = Arow0 + (size_t)16 * KP;
    for (int k0 = 0; k0 < KP; k0 += 32) {
        if (k0 + 64 < KP) {
            __builtin_prefetch((const void*)(Arow0 + k0 + 64), 0, 3);
            __builtin_prefetch((const void*)(Arow1 + k0 + 64), 0, 3);
        }
        v16h af0 = load_a_frag(Arow0, k0, half16);
        v16h af1 = load_a_frag(Arow1, k0, half16);
        #pragma unroll
        for (int j = 0; j < 4; ++j) {
            v16h bf = load_b_frag(W, KP, n0 + j * 16 + l16, k0, half16);
            acc[0][j] = wmma_f16(af0, bf, acc[0][j]);
            acc[1][j] = wmma_f16(af1, bf, acc[1][j]);
        }
    }
    const float qscale = 0.1414213562373095f; // 50^-0.5
    #pragma unroll
    for (int a = 0; a < 2; ++a)
        #pragma unroll
        for (int j = 0; j < 4; ++j) {
            int n = n0 + j * 16 + l16;
            if (n >= 3 * Dn) continue;
            int sect, idx;
            if (n < Dn)           { sect = 0; idx = n; }
            else if (n < 2 * Dn)  { sect = 1; idx = n - Dn; }
            else                  { sect = 2; idx = n - 2 * Dn; }
            int h = idx / HDn, d = idx - h * HDn;
            #pragma unroll
            for (int r = 0; r < 8; ++r) {
                int m = m0 + a * 16 + r + 8 * half16;
                int b = m >> 9, t = m & (Tn - 1);
                size_t bh = (size_t)(b * Hn + h);
                float v = acc[a][j][r] + bias[n];
                if (sect == 0)      qh[(bh * Tn + t) * HDP + d] = (_Float16)(v * qscale);
                else if (sect == 1) kh[(bh * Tn + t) * HDP + d] = (_Float16)v;
                else                vth[bh * (HDP * Tn) + (size_t)d * Tn + t] = (_Float16)v;
            }
        }
}

// ---------------- fused scores + masked softmax ----------------------------
// One block = 16 query rows x all 512 keys for one (b,h). 4 waves x 128 cols.
__global__ __launch_bounds__(128)
void k_scores_softmax(const _Float16* __restrict__ qh, const _Float16* __restrict__ kh,
                      const unsigned char* __restrict__ mask, _Float16* __restrict__ P) {
    const int blk = blockIdx.x;        // BH * 32
    const int bh = blk >> 5, mt = blk & 31;
    const int b = bh / Hn;
    const int lane = threadIdx.x & 31, wave = threadIdx.x >> 5;
    const int half16 = lane >> 4, l16 = lane & 15;
    const int m0 = mt * 16;
    const _Float16* Q = qh + (size_t)bh * Tn * HDP;
    const _Float16* K = kh + (size_t)bh * Tn * HDP;
    v8f acc[8];
    #pragma unroll
    for (int j = 0; j < 8; ++j)
        #pragma unroll
        for (int r = 0; r < 8; ++r) acc[j][r] = 0.f;
    const _Float16* Qrow = Q + (size_t)(m0 + l16) * HDP;
    #pragma unroll
    for (int k0 = 0; k0 < HDP; k0 += 32) {
        v16h af = load_a_frag(Qrow, k0, half16);
        #pragma unroll
        for (int j = 0; j < 8; ++j) {
            v16h bf = load_b_frag(K, HDP, wave * 128 + j * 16 + l16, k0, half16);
            acc[j] = wmma_f16(af, bf, acc[j]);
        }
    }
    // key mask -> -inf
    #pragma unroll
    for (int j = 0; j < 8; ++j) {
        int n = wave * 128 + j * 16 + l16;
        if (mask[b * Tn + n]) {
            #pragma unroll
            for (int r = 0; r < 8; ++r) acc[j][r] = -__builtin_inff();
        }
    }
    __shared__ float red[4][16];
    float rowmax[8], rowsum[8];
    #pragma unroll
    for (int r = 0; r < 8; ++r) {
        float mx = -__builtin_inff();
        #pragma unroll
        for (int j = 0; j < 8; ++j) mx = fmaxf(mx, acc[j][r]);
        mx = hmax16(mx);
        if (l16 == 0) red[wave][half16 * 8 + r] = mx;
        __syncthreads();
        rowmax[r] = fmaxf(fmaxf(red[0][half16 * 8 + r], red[1][half16 * 8 + r]),
                          fmaxf(red[2][half16 * 8 + r], red[3][half16 * 8 + r]));
        __syncthreads();
    }
    #pragma unroll
    for (int r = 0; r < 8; ++r) {
        float s = 0.f;
        #pragma unroll
        for (int j = 0; j < 8; ++j) {
            float e = __expf(acc[j][r] - rowmax[r]);
            acc[j][r] = e;
            s += e;
        }
        s = hsum16(s);
        if (l16 == 0) red[wave][half16 * 8 + r] = s;
        __syncthreads();
        rowsum[r] = red[0][half16 * 8 + r] + red[1][half16 * 8 + r] +
                    red[2][half16 * 8 + r] + red[3][half16 * 8 + r];
        __syncthreads();
    }
    _Float16* Pp = P + (size_t)bh * Tn * Tn;
    #pragma unroll
    for (int j = 0; j < 8; ++j) {
        int n = wave * 128 + j * 16 + l16;
        #pragma unroll
        for (int r = 0; r < 8; ++r) {
            int m = m0 + r + 8 * half16;
            Pp[(size_t)m * Tn + n] = (_Float16)(acc[j][r] / rowsum[r]);
        }
    }
}

// ---------------- attn = P @ V : [512,512] x [512,64] per (b,h) ------------
// Block tile 128x64, wave tile 32x64.
__global__ __launch_bounds__(128)
void k_attn_v(const _Float16* __restrict__ P, const _Float16* __restrict__ vth,
              _Float16* __restrict__ attnH) {
    const int blk = blockIdx.x;        // BH * 4
    const int bh = blk >> 2, mt = blk & 3;
    const int b = bh / Hn, h = bh - b * Hn;
    const int lane = threadIdx.x & 31, wave = threadIdx.x >> 5;
    const int half16 = lane >> 4, l16 = lane & 15;
    const int m0 = mt * 128 + wave * 32;
    const _Float16* Pp = P + (size_t)bh * Tn * Tn;
    const _Float16* V = vth + (size_t)bh * HDP * Tn;   // [64,512] rows = d
    v8f acc[2][4];
    #pragma unroll
    for (int a = 0; a < 2; ++a)
        #pragma unroll
        for (int j = 0; j < 4; ++j)
            #pragma unroll
            for (int r = 0; r < 8; ++r) acc[a][j][r] = 0.f;
    const _Float16* Arow0 = Pp + (size_t)(m0 + l16) * Tn;
    const _Float16* Arow1 = Arow0 + (size_t)16 * Tn;
    for (int k0 = 0; k0 < Tn; k0 += 32) {
        if (k0 + 64 < Tn) {
            __builtin_prefetch((const void*)(Arow0 + k0 + 64), 0, 3);
            __builtin_prefetch((const void*)(Arow1 + k0 + 64), 0, 3);
        }
        v16h af0 = load_a_frag(Arow0, k0, half16);
        v16h af1 = load_a_frag(Arow1, k0, half16);
        #pragma unroll
        for (int j = 0; j < 4; ++j) {
            v16h bf = load_b_frag(V, Tn, j * 16 + l16, k0, half16);
            acc[0][j] = wmma_f16(af0, bf, acc[0][j]);
            acc[1][j] = wmma_f16(af1, bf, acc[1][j]);
        }
    }
    #pragma unroll
    for (int a = 0; a < 2; ++a)
        #pragma unroll
        for (int j = 0; j < 4; ++j) {
            int d = j * 16 + l16;
            if (d >= HDn) continue;
            #pragma unroll
            for (int r = 0; r < 8; ++r) {
                int t = m0 + a * 16 + r + 8 * half16;
                attnH[((size_t)(b * Tn + t)) * KP + h * HDn + d] = (_Float16)acc[a][j][r];
            }
        }
}

// ---------------------------------------------------------------------------

extern "C" void kernel_launch(void* const* d_in, const int* in_sizes, int n_in,
                              void* d_out, int out_size, void* d_ws, size_t ws_size,
                              hipStream_t stream) {
    (void)in_sizes; (void)n_in; (void)out_size; (void)ws_size;
    const int*   ori    = (const int*)d_in[0];
    const float* x      = (const float*)d_in[1];
    const unsigned char* xmask = (const unsigned char*)d_in[2];
    const float *dw[4], *db[4], *pw[4], *pb[4];
    for (int i = 0; i < 4; ++i) {
        dw[i] = (const float*)d_in[3 + 4 * i];
        db[i] = (const float*)d_in[4 + 4 * i];
        pw[i] = (const float*)d_in[5 + 4 * i];
        pb[i] = (const float*)d_in[6 + 4 * i];
    }
    const float* in_w  = (const float*)d_in[19];
    const float* in_b  = (const float*)d_in[20];
    const float* out_w = (const float*)d_in[21];
    const float* out_b = (const float*)d_in[22];
    const float* ffc_w = (const float*)d_in[23];
    const float* ffc_b = (const float*)d_in[24];
    float* out = (float*)d_out;

    // ---- workspace layout (bytes) ----
    char* ws = (char*)d_ws;
    const size_t szA   = (size_t)Mn * Dn * 4;          // 32,768,000
    const size_t szH   = (size_t)Mn * KP * 2;          // 16,777,216
    const size_t szQKH = (size_t)BH * Tn * HDP * 2;    // 20,971,520
    const size_t szW   = (size_t)KP * KP * 2;          // 524,288 (512 rows x 512)
    const size_t szWin = (size_t)1536 * KP * 2;        // 1,572,864
    const size_t szP   = (size_t)BH * Tn * Tn * 2;     // 167,772,160

    float*    A0    = (float*)ws;                 size_t off = szA;
    float*    A1    = (float*)(ws + off);         off += szA;
    const size_t zstart = off;
    _Float16* yh    = (_Float16*)(ws + off);      off += szH;
    _Float16* actH  = (_Float16*)(ws + off);      off += szH;
    _Float16* attnH = (_Float16*)(ws + off);      off += szH;
    _Float16* qh    = (_Float16*)(ws + off);      off += szQKH;
    _Float16* kh    = (_Float16*)(ws + off);      off += szQKH;
    _Float16* vth   = (_Float16*)(ws + off);      off += szQKH;
    _Float16* pwh[4];
    for (int i = 0; i < 4; ++i) { pwh[i] = (_Float16*)(ws + off); off += szW; }
    _Float16* inWh  = (_Float16*)(ws + off);      off += szWin;
    _Float16* outWh = (_Float16*)(ws + off);      off += szW;
    _Float16* ffcWh = (_Float16*)(ws + off);      off += szW;
    const size_t zend = off;
    _Float16* P     = (_Float16*)(ws + off);      off += szP;

    // ---- zero pad-sensitive f16 region (every call; ws is poisoned) ----
    {
        size_t nwords = (zend - zstart) / 4;
        k_zero_words<<<4096, 256, 0, stream>>>((uint32_t*)(ws + zstart), nwords);
    }
    // ---- pack weights to f16 [Npad,512] ----
    for (int i = 0; i < 4; ++i)
        k_pack_w<<<(Dn * KP + 255) / 256, 256, 0, stream>>>(pw[i], pwh[i], Dn);
    k_pack_w<<<(3 * Dn * KP + 255) / 256, 256, 0, stream>>>(in_w, inWh, 3 * Dn);
    k_pack_w<<<(Dn * KP + 255) / 256, 256, 0, stream>>>(out_w, outWh, Dn);
    k_pack_w<<<(Dn * KP + 255) / 256, 256, 0, stream>>>(ffc_w, ffcWh, Dn);

    const size_t nElem = (size_t)Mn * Dn;
    const int ETHR = 256;
    const int EBLK = (int)((nElem + ETHR - 1) / ETHR);

    // ---- pos-emb + x -> A0 ----
    k_posemb<<<EBLK, ETHR, 0, stream>>>(ori, x, A0);

    // ---- 4 separable-conv layers with residual ----
    dim3 gemmGrid(Mn / 128, (Dn + 63) / 64);   // 128 x 8
    float* cur = A0; float* nxt = A1;
    for (int i = 0; i < 4; ++i) {
        k_dwconv<<<EBLK, ETHR, 0, stream>>>(cur, dw[i], db[i], yh);
        k_gemm_f32out<<<gemmGrid, 128, 0, stream>>>(
            yh, pwh[i], pb[i], cur, nxt, (i == 3) ? actH : (_Float16*)nullptr, Dn);
        float* tmp = cur; cur = nxt; nxt = tmp;
    }
    // out3: f32 in `cur`, f16 in actH

    // ---- QKV projection (M x 1536 x 512) ----
    dim3 qkvGrid(Mn / 128, 1536 / 64);         // 128 x 24
    k_gemm_qkv<<<qkvGrid, 128, 0, stream>>>(actH, inWh, in_b, qh, kh, vth);

    // ---- fused scores + masked softmax -> P ----
    k_scores_softmax<<<BH * (Tn / 16), 128, 0, stream>>>(qh, kh, xmask, P);

    // ---- attn = P @ V -> attnH (heads concatenated, [M,512] f16) ----
    k_attn_v<<<BH * (Tn / 128), 128, 0, stream>>>(P, vth, attnH);

    // ---- out-proj + residual(out3) -> nxt (f32) and actH (f16) ----
    k_gemm_f32out<<<gemmGrid, 128, 0, stream>>>(attnH, outWh, out_b, cur, nxt, actH, Dn);

    // ---- FFC + residual -> d_out ----
    k_gemm_f32out<<<gemmGrid, 128, 0, stream>>>(actH, ffcWh, ffc_b, nxt, out,
                                                (_Float16*)nullptr, Dn);
}
